// DecoderRNN_46789373723287
// MI455X (gfx1250) — matrix operand order repaired and checked by reading, not compiled, not run on hardware
//
#include <hip/hip_runtime.h>
#include <hip/hip_bf16.h>

// Problem constants (match reference).
constexpr int H = 1024;
constexpr int O = 8192;
constexpr int L = 2;
constexpr int T = 64;
constexpr int B = 256;

typedef __attribute__((ext_vector_type(16))) __bf16 v16bf;
typedef __attribute__((ext_vector_type(8)))  __bf16 v8bf;
typedef __attribute__((ext_vector_type(8)))  float  v8f;

// ---------------------------------------------------------------------------
// Operand loaders for v_wmma_f32_16x16x32_bf16 (wave32).
//
// A (16x32, MxK): lanes 0-15 hold row M=lane with K = {k0..k0+7, k0+16..k0+23};
//                 lanes 16-31 hold row M=lane-16 with K = {k0+8..15, k0+24..31}.
// B (32x16, KxN): lanes 0-15 hold col N=lane with K = {k0..k0+7, k0+16..k0+23};
//                 lanes 16-31 hold col N=lane-16 with K = {k0+8..15, k0+24..31}.
// Both GEMMs are X @ W^T with W row-major [out,in], so the B operand is a
// contiguous row of W along K — no transpose loads needed.
// ---------------------------------------------------------------------------
__device__ __forceinline__ v16bf load_a(const __bf16* __restrict__ row, int k0, int lane) {
  const int sub = (lane >> 4) << 3;                       // 0 or 8
  v8bf lo = *(const v8bf*)(row + k0 + sub);
  v8bf hi = *(const v8bf*)(row + k0 + 16 + sub);
  union { v16bf v; v8bf h[2]; } u;
  u.h[0] = lo; u.h[1] = hi;
  return u.v;
}

__device__ __forceinline__ v16bf load_b(const __bf16* __restrict__ row, int k0, int lane) {
  const int sub = (lane >> 4) << 3;                       // 0 or 8
  v8bf lo = *(const v8bf*)(row + k0 + sub);
  v8bf hi = *(const v8bf*)(row + k0 + 16 + sub);
  union { v16bf v; v8bf h[2]; } u;
  u.h[0] = lo; u.h[1] = hi;
  return u.v;
}

__device__ __forceinline__ v8f wmma_bf16(v16bf a, v16bf b, v8f c) {
  return __builtin_amdgcn_wmma_f32_16x16x32_bf16(false, a, false, b, (short)0, c,
                                                 false, false);
}

// ---------------------------------------------------------------------------
// Phase 0: fp32 -> bf16 conversion, bias fusion.
// ---------------------------------------------------------------------------
__global__ void __launch_bounds__(256)
cvt_f32_bf16(const float* __restrict__ in, __bf16* __restrict__ out, int n) {
  int i = blockIdx.x * 256 + threadIdx.x;
  if (i < n) out[i] = (__bf16)in[i];
}

__global__ void __launch_bounds__(256)
bias_fuse(const float* __restrict__ bih, const float* __restrict__ bhh,
          float* __restrict__ out, int n) {
  int i = blockIdx.x * 256 + threadIdx.x;
  if (i < n) out[i] = bih[i] + bhh[i];
}

// ---------------------------------------------------------------------------
// Phase 1: one RNN layer for one time step (launched 2*T times; stream order
// realizes the recurrence).
//   Out[m,n] = tanh( sum_k X[m,k]*Wih[n,k] + sum_k Hp[m,k]*Whh[n,k] + bias[n] )
// One wave per 16x16 output tile: 1024 waves; this phase is launch/latency
// bound, so maximize wave parallelism (activations + weights are L2-resident).
// ---------------------------------------------------------------------------
__global__ void __launch_bounds__(256, 1)
rnn_layer(const __bf16* __restrict__ X, const __bf16* __restrict__ Hp,
          const __bf16* __restrict__ Wih, const __bf16* __restrict__ Whh,
          const float* __restrict__ bias, __bf16* __restrict__ Out) {
  const int lane = threadIdx.x & 31;
  const int tile = blockIdx.x * 8 + (threadIdx.x >> 5);   // 0..1023
  const int m0 = (tile >> 6) << 4;                        // 16 M-tiles
  const int n0 = (tile & 63) << 4;                        // 64 N-tiles
  const int l15 = lane & 15;

  const __bf16* xrow  = X   + (m0 + l15) * H;
  const __bf16* hrow  = Hp  + (m0 + l15) * H;
  const __bf16* w1row = Wih + (n0 + l15) * H;
  const __bf16* w2row = Whh + (n0 + l15) * H;

  v8f acc = {};
  for (int k0 = 0; k0 < H; k0 += 32) {
    v16bf a1 = load_a(xrow, k0, lane);
    v16bf b1 = load_b(w1row, k0, lane);
    v16bf a2 = load_a(hrow, k0, lane);
    v16bf b2 = load_b(w2row, k0, lane);
    acc = wmma_bf16(a1, b1, acc);
    acc = wmma_bf16(a2, b2, acc);
  }

  const int n = n0 + l15;
  const float bv = bias[n];
  const int mbase = m0 + ((lane >> 4) << 3);
  #pragma unroll
  for (int r = 0; r < 8; ++r) {
    float v = tanhf(acc[r] + bv);
    Out[(mbase + r) * H + n] = (__bf16)v;
  }
}

// ---------------------------------------------------------------------------
// Phase 2: batched FC over all steps: logits[m,n] = A[m,:]·Wfc[n,:] + fcb[n],
// M = T*B = 16384, N = O = 8192, K = H = 1024.
//
// 4x4 register blocking: one wave owns a 64x64 macro-tile (16 v8f accums =
// 128 VGPRs — cheap under CDNA5's 1024-VGPR/wave budget; launch_bounds(256,1)
// tells the allocator not to cap registers for occupancy, so the accumulators
// never spill). Per K-iter: 8 operand loads (256 B/lane) feed 16 WMMAs — 2x
// the arithmetic intensity of a 2x2 tiling, keeping operand bandwidth within
// L1/L2 reach at WMMA rates. 256 x 128 macro-tiles = 32768 waves. Epilogue
// stores are non-temporal so the 512 MB one-shot output doesn't evict the
// L2-resident weights.
// ---------------------------------------------------------------------------
__global__ void __launch_bounds__(256, 1)
fc_gemm(const __bf16* __restrict__ A, const __bf16* __restrict__ Wfc,
        const float* __restrict__ fcb, float* __restrict__ Outp) {
  const int lane = threadIdx.x & 31;
  const int w = blockIdx.x * 8 + (threadIdx.x >> 5);      // 0..32767
  const int m0 = (w >> 7) << 6;                           // 256 M macro-tiles
  const int n0 = (w & 127) << 6;                          // 128 N macro-tiles
  const int l15 = lane & 15;

  const __bf16* ar0 = A   + (m0      + l15) * H;
  const __bf16* ar1 = A   + (m0 + 16 + l15) * H;
  const __bf16* ar2 = A   + (m0 + 32 + l15) * H;
  const __bf16* ar3 = A   + (m0 + 48 + l15) * H;
  const __bf16* br0 = Wfc + (n0      + l15) * H;
  const __bf16* br1 = Wfc + (n0 + 16 + l15) * H;
  const __bf16* br2 = Wfc + (n0 + 32 + l15) * H;
  const __bf16* br3 = Wfc + (n0 + 48 + l15) * H;

  v8f acc[4][4] = {};
  for (int k0 = 0; k0 < H; k0 += 32) {
    v16bf a0 = load_a(ar0, k0, lane);
    v16bf a1 = load_a(ar1, k0, lane);
    v16bf a2 = load_a(ar2, k0, lane);
    v16bf a3 = load_a(ar3, k0, lane);
    v16bf b0 = load_b(br0, k0, lane);
    v16bf b1 = load_b(br1, k0, lane);
    v16bf b2 = load_b(br2, k0, lane);
    v16bf b3 = load_b(br3, k0, lane);

    acc[0][0] = wmma_bf16(a0, b0, acc[0][0]);
    acc[0][1] = wmma_bf16(a0, b1, acc[0][1]);
    acc[0][2] = wmma_bf16(a0, b2, acc[0][2]);
    acc[0][3] = wmma_bf16(a0, b3, acc[0][3]);
    acc[1][0] = wmma_bf16(a1, b0, acc[1][0]);
    acc[1][1] = wmma_bf16(a1, b1, acc[1][1]);
    acc[1][2] = wmma_bf16(a1, b2, acc[1][2]);
    acc[1][3] = wmma_bf16(a1, b3, acc[1][3]);
    acc[2][0] = wmma_bf16(a2, b0, acc[2][0]);
    acc[2][1] = wmma_bf16(a2, b1, acc[2][1]);
    acc[2][2] = wmma_bf16(a2, b2, acc[2][2]);
    acc[2][3] = wmma_bf16(a2, b3, acc[2][3]);
    acc[3][0] = wmma_bf16(a3, b0, acc[3][0]);
    acc[3][1] = wmma_bf16(a3, b1, acc[3][1]);
    acc[3][2] = wmma_bf16(a3, b2, acc[3][2]);
    acc[3][3] = wmma_bf16(a3, b3, acc[3][3]);
  }

  float bb[4];
  #pragma unroll
  for (int j = 0; j < 4; ++j) bb[j] = fcb[n0 + 16 * j + l15];
  const int mbase = m0 + ((lane >> 4) << 3);
  #pragma unroll
  for (int i = 0; i < 4; ++i) {
    #pragma unroll
    for (int r = 0; r < 8; ++r) {
      size_t rowOff = (size_t)(mbase + 16 * i + r) * O;
      #pragma unroll
      for (int j = 0; j < 4; ++j) {
        __builtin_nontemporal_store(acc[i][j][r] + bb[j],
                                    Outp + rowOff + n0 + 16 * j + l15);
      }
    }
  }
}

// ---------------------------------------------------------------------------
// Host launcher.
// Input order: x, hidden, embedded, W_ih, W_hh, b_ih, b_hh, fc_W, fc_b.
// ---------------------------------------------------------------------------
extern "C" void kernel_launch(void* const* d_in, const int* in_sizes, int n_in,
                              void* d_out, int out_size, void* d_ws, size_t ws_size,
                              hipStream_t stream) {
  const float* x_f    = (const float*)d_in[0];   // [1,B,H]
  const float* hid_f  = (const float*)d_in[1];   // [L,B,H]
  const float* Wih_f  = (const float*)d_in[3];   // [L,H,H]
  const float* Whh_f  = (const float*)d_in[4];   // [L,H,H]
  const float* bih_f  = (const float*)d_in[5];   // [L,H]
  const float* bhh_f  = (const float*)d_in[6];   // [L,H]
  const float* fcW_f  = (const float*)d_in[7];   // [O,H]
  const float* fcb_f  = (const float*)d_in[8];   // [O]
  float* out = (float*)d_out;                    // [T,1,B,O]

  // Workspace carve-up (all chunk sizes are multiples of 256 bytes).
  char* p = (char*)d_ws;
  __bf16* wih_bf  = (__bf16*)p; p += (size_t)L * H * H * 2;   // 4 MiB
  __bf16* whh_bf  = (__bf16*)p; p += (size_t)L * H * H * 2;   // 4 MiB
  __bf16* fcw_bf  = (__bf16*)p; p += (size_t)O * H * 2;       // 16 MiB
  float*  bias    = (float*)p;  p += (size_t)L * H * 4;       // 8 KiB
  __bf16* x0_bf   = (__bf16*)p; p += (size_t)B * H * 2;       // 512 KiB
  __bf16* hin_bf  = (__bf16*)p; p += (size_t)L * B * H * 2;   // 1 MiB
  __bf16* l0_buf  = (__bf16*)p; p += (size_t)2 * B * H * 2;   // 1 MiB ping-pong
  __bf16* out_buf = (__bf16*)p; p += (size_t)T * B * H * 2;   // 32 MiB

  const int BH = B * H;
  auto cvt = [&](const float* src, __bf16* dst, int n) {
    cvt_f32_bf16<<<(n + 255) / 256, 256, 0, stream>>>(src, dst, n);
  };

  // Phase 0: weight/state conversion + bias fusion.
  cvt(Wih_f, wih_bf, L * H * H);
  cvt(Whh_f, whh_bf, L * H * H);
  cvt(fcW_f, fcw_bf, O * H);
  cvt(x_f,   x0_bf,  BH);
  cvt(hid_f, hin_bf, L * BH);
  bias_fuse<<<(L * H + 255) / 256, 256, 0, stream>>>(bih_f, bhh_f, bias, L * H);

  // Phase 1: sequential recurrence, 2 layer-kernels per step.
  // l0_buf ping-pongs layer-0 hidden/output; out_buf[t] doubles as layer-1
  // hidden for step t+1 and as the phase-2 GEMM input.
  for (int t = 0; t < T; ++t) {
    const __bf16* x0  = (t == 0) ? x0_bf  : out_buf + (size_t)(t - 1) * BH;
    const __bf16* h0  = (t == 0) ? hin_bf : l0_buf + (size_t)((t - 1) & 1) * BH;
    __bf16*       o0  = l0_buf + (size_t)(t & 1) * BH;
    rnn_layer<<<128, 256, 0, stream>>>(x0, h0, wih_bf, whh_bf, bias, o0);

    const __bf16* h1  = (t == 0) ? hin_bf + BH : out_buf + (size_t)(t - 1) * BH;
    __bf16*       o1  = out_buf + (size_t)t * BH;
    rnn_layer<<<128, 256, 0, stream>>>(o0, h1, wih_bf + H * H, whh_bf + H * H,
                                       bias + H, o1);
  }

  // Phase 2: one big GEMM for all logits: [T*B, H] x [H, O].
  fc_gemm<<<4096, 256, 0, stream>>>(out_buf, fcw_bf, fcb_f, out);
}
